// SimpleMoE_10960756539443
// MI455X (gfx1250) — matrix operand order repaired and checked by reading, not compiled, not run on hardware
//
#include <hip/hip_runtime.h>
#include <hip/hip_bf16.h>

// ---------------- problem constants ----------------
#define B_TOK 2048
#define SEQ   128
#define DIM   1024
#define NEXP  16
#define HID   2048
#define NROWS (B_TOK*2)   // top-2 routing: exactly 2 rows per token

typedef __attribute__((ext_vector_type(16))) __bf16 v16bf;
typedef __attribute__((ext_vector_type(8)))  float  v8f;

union FragBF { v16bf v; unsigned u[8]; };

// round-to-nearest-even f32 -> bf16
__device__ __forceinline__ unsigned short f2bf(float f) {
  unsigned u = __float_as_uint(f);
  unsigned r = u + 0x7FFFu + ((u >> 16) & 1u);
  return (unsigned short)(r >> 16);
}
__device__ __forceinline__ unsigned packbf(float lo, float hi) {
  return (unsigned)f2bf(lo) | ((unsigned)f2bf(hi) << 16);
}

// ---------------- workspace layout (bytes) ----------------
constexpr size_t OFF_H    = 0;                                   // [B,D] f32
constexpr size_t OFF_HB   = OFF_H   + (size_t)B_TOK*DIM*4;       // [B,D] bf16
constexpr size_t OFF_TKV  = OFF_HB  + (size_t)B_TOK*DIM*2;       // [B,2] f32
constexpr size_t OFF_TKI  = OFF_TKV + (size_t)B_TOK*2*4;         // [B,2] i32
constexpr size_t OFF_CNT  = OFF_TKI + (size_t)B_TOK*2*4;         // [16] i32
constexpr size_t OFF_BASE = OFF_CNT + 256;                       // [16] i32
constexpr size_t OFF_CUR  = OFF_BASE + 256;                      // [16] i32
constexpr size_t OFF_RL   = OFF_CUR + 256;                       // [4096] i32 (token*2+k)
constexpr size_t OFF_ACT  = OFF_RL  + (size_t)NROWS*4;           // [4096,H] bf16
constexpr size_t OFF_Y    = OFF_ACT + (size_t)NROWS*HID*2;       // [B,2,D] f32
constexpr size_t WS_MIN   = OFF_Y   + (size_t)B_TOK*2*DIM*4;     // ~46.2 MB
// optional packed-bf16 weight copies (B-fragment pair order):
constexpr size_t OFF_W1P  = WS_MIN;                              // [E][D/2][H] u32
constexpr size_t OFF_W2P  = OFF_W1P + (size_t)NEXP*(DIM/2)*HID*4;// [E][H/2][D] u32
constexpr size_t WS_FULL  = OFF_W2P + (size_t)NEXP*(HID/2)*DIM*4;// ~180.5 MB

// ---------------- kernel 0: zero expert counts ----------------
__global__ void moe_zero_counts(int* counts) {
  if (threadIdx.x < NEXP) counts[threadIdx.x] = 0;
}

// ---------------- kernel 1: embedding mean-pool -> h (f32) + hb (bf16) ----------------
__global__ __launch_bounds__(256) void moe_embed_pool(
    const int* __restrict__ x, const float* __restrict__ embed,
    float* __restrict__ h, unsigned short* __restrict__ hb) {
  const int b = blockIdx.x, t = threadIdx.x;    // t covers D/4 float4 lanes
  const int* xr = x + (size_t)b * SEQ;
  float4 acc = {0.f, 0.f, 0.f, 0.f};
  for (int s = 0; s < SEQ; ++s) {
    const float4 v = ((const float4*)(embed + (size_t)xr[s] * DIM))[t];
    acc.x += v.x; acc.y += v.y; acc.z += v.z; acc.w += v.w;
  }
  const float inv = 1.0f / (float)SEQ;
  acc.x *= inv; acc.y *= inv; acc.z *= inv; acc.w *= inv;
  ((float4*)(h + (size_t)b * DIM))[t] = acc;
  uint2 p; p.x = packbf(acc.x, acc.y); p.y = packbf(acc.z, acc.w);
  ((uint2*)(hb + (size_t)b * DIM))[t] = p;
}

// ---------------- kernel 2: gate -> softmax -> top-2 ----------------
__global__ __launch_bounds__(256) void moe_gate_topk(
    const float* __restrict__ h, const float* __restrict__ gate_w,
    const float* __restrict__ gate_b, float* __restrict__ tkval,
    int* __restrict__ tkidx, int* __restrict__ counts, int* __restrict__ out_idx) {
  const int b = blockIdx.x, t = threadIdx.x;
  const int e = t & 15, j = t >> 4;
  __shared__ float part[256];
  __shared__ float sc[NEXP];
  const float* hr = h + (size_t)b * DIM;
  float acc = 0.f;
  for (int d = j; d < DIM; d += 16) acc += hr[d] * gate_w[d * NEXP + e];
  part[j * 16 + e] = acc;
  __syncthreads();
  if (t < NEXP) {
    float s = gate_b[t];
    for (int jj = 0; jj < 16; ++jj) s += part[jj * 16 + t];
    sc[t] = s;
  }
  __syncthreads();
  if (t == 0) {
    float m = sc[0];
    for (int i = 1; i < NEXP; ++i) m = fmaxf(m, sc[i]);
    float ex[NEXP]; float sum = 0.f;
    for (int i = 0; i < NEXP; ++i) { ex[i] = expf(sc[i] - m); sum += ex[i]; }
    const float inv = 1.0f / sum;
    int i0 = 0; float v0 = ex[0];
    for (int i = 1; i < NEXP; ++i) if (ex[i] > v0) { v0 = ex[i]; i0 = i; }   // ties -> lower idx
    int i1 = -1; float v1 = -1.f;
    for (int i = 0; i < NEXP; ++i) if (i != i0 && ex[i] > v1) { v1 = ex[i]; i1 = i; }
    tkval[b * 2 + 0] = v0 * inv; tkval[b * 2 + 1] = v1 * inv;
    tkidx[b * 2 + 0] = i0;       tkidx[b * 2 + 1] = i1;
    out_idx[b * 2 + 0] = i0;     out_idx[b * 2 + 1] = i1;
    atomicAdd(&counts[i0], 1);   atomicAdd(&counts[i1], 1);
  }
}

// ---------------- kernel 3: exclusive prefix of counts ----------------
__global__ void moe_prefix(const int* __restrict__ counts, int* __restrict__ base,
                           int* __restrict__ cursor) {
  if (threadIdx.x == 0) {
    int acc = 0;
    for (int e = 0; e < NEXP; ++e) { base[e] = acc; acc += counts[e]; cursor[e] = 0; }
  }
}

// ---------------- kernel 4: slot assignment ----------------
__global__ __launch_bounds__(256) void moe_assign(
    const int* __restrict__ tkidx, const int* __restrict__ base,
    int* __restrict__ cursor, int* __restrict__ rowlist) {
  const int b = blockIdx.x * blockDim.x + threadIdx.x;
  if (b >= B_TOK) return;
  for (int k = 0; k < 2; ++k) {
    const int e = tkidx[b * 2 + k];
    const int slot = atomicAdd(&cursor[e], 1);
    rowlist[base[e] + slot] = b * 2 + k;
  }
}

// ------------- kernel 4b: repack weights f32 -> bf16 pair-packed (B-frag order) -------
// src: [E][K][N] f32 ; dst: [E][K/2][N] u32 where dst = pack(src[2k2][n], src[2k2+1][n])
__global__ __launch_bounds__(256) void moe_pack_w(
    const float* __restrict__ src, unsigned* __restrict__ dst, int K, int N) {
  const size_t i = (size_t)blockIdx.x * 256 + threadIdx.x;
  const size_t total = (size_t)NEXP * (K / 2) * N;
  if (i >= total) return;
  const size_t col  = i % (size_t)N;
  const size_t rest = i / (size_t)N;      // e*(K/2) + k2  ->  2*rest = e*K + 2*k2
  const size_t row0 = rest * 2;
  dst[i] = packbf(src[row0 * N + col], src[(row0 + 1) * N + col]);
}

// ---------------- FFN GEMM cores (bf16 WMMA, M=64 x N=128 per block) -----------------
// block = 256 thr (8 waves), one 16-col N-subtile per wave, K loop of 32.
template <bool PACKED>
__global__ __launch_bounds__(256) void moe_ffn1_wmma(
    const unsigned short* __restrict__ hb, const float* __restrict__ w1,
    const unsigned* __restrict__ w1p, const float* __restrict__ b1,
    const int* __restrict__ counts, const int* __restrict__ base,
    const int* __restrict__ rowlist, unsigned short* __restrict__ act) {
  const int e = blockIdx.z;
  const int count = counts[e];
  const int tileBase = blockIdx.y * 64;
  if (tileBase >= count) return;
  const int baseE = base[e];
  const int t = threadIdx.x;
  const int wave = t >> 5, lane = t & 31, half = lane >> 4, id = lane & 15;
  const int col = blockIdx.x * 128 + wave * 16 + id;

  __shared__ unsigned AsU[64 * 18];      // 64 rows x 32 bf16, stride 18 dwords (conflict-free)
  v8f acc[4] = {};

  // cooperative A-tile load mapping: 4 threads per row, 8 bf16 (16B) each
  const int r = t >> 2, cofs = (t & 3) * 8;
  const bool rok = (tileBase + r) < count;
  const int tok = rok ? (rowlist[baseE + tileBase + r] >> 1) : 0;
  const float*    wbase = w1  + (size_t)e * DIM * HID;
  const unsigned* wpk   = w1p + (size_t)e * (DIM / 2) * HID;

  for (int kb = 0; kb < DIM; kb += 32) {
    __syncthreads();
    uint4 av = {0u, 0u, 0u, 0u};
    if (rok) av = *(const uint4*)(hb + (size_t)tok * DIM + kb + cofs);
    unsigned* dst = &AsU[r * 18 + (cofs >> 1)];
    dst[0] = av.x; dst[1] = av.y; dst[2] = av.z; dst[3] = av.w;
    __syncthreads();

    // B fragment: 32x16 bf16, lane id = column
    FragBF bf;
#pragma unroll
    for (int v = 0; v < 8; ++v) {
      if constexpr (PACKED) {
        const int k2 = (kb >> 1) + v + half * 8;
        bf.u[v] = wpk[(size_t)k2 * HID + col];
      } else {
        const int k = kb + v * 2 + half * 16;
        bf.u[v] = packbf(wbase[(size_t)k * HID + col], wbase[(size_t)(k + 1) * HID + col]);
      }
    }
#pragma unroll
    for (int m = 0; m < 4; ++m) {
      FragBF af;
      const unsigned* src = &AsU[(m * 16 + id) * 18 + half * 8];
#pragma unroll
      for (int v = 0; v < 8; ++v) af.u[v] = src[v];
      acc[m] = __builtin_amdgcn_wmma_f32_16x16x32_bf16(
          false, af.v, false, bf.v, (short)0, acc[m], false, false);
    }
  }

  const float bias = b1[e * HID + col];
#pragma unroll
  for (int m = 0; m < 4; ++m) {
#pragma unroll
    for (int jj = 0; jj < 8; ++jj) {
      const int lr = m * 16 + jj + half * 8;
      if (tileBase + lr < count) {
        const float vv = fmaxf(acc[m][jj] + bias, 0.f);
        act[(size_t)(baseE + tileBase + lr) * HID + col] = f2bf(vv);
      }
    }
  }
}

template <bool PACKED>
__global__ __launch_bounds__(256) void moe_ffn2_wmma(
    const unsigned short* __restrict__ act, const float* __restrict__ w2,
    const unsigned* __restrict__ w2p, const float* __restrict__ b2,
    const float* __restrict__ tkval, const int* __restrict__ counts,
    const int* __restrict__ base, const int* __restrict__ rowlist,
    float* __restrict__ ybuf) {
  const int e = blockIdx.z;
  const int count = counts[e];
  const int tileBase = blockIdx.y * 64;
  if (tileBase >= count) return;
  const int baseE = base[e];
  const int t = threadIdx.x;
  const int wave = t >> 5, lane = t & 31, half = lane >> 4, id = lane & 15;
  const int col = blockIdx.x * 128 + wave * 16 + id;

  __shared__ unsigned AsU[64 * 18];
  v8f acc[4] = {};

  const int r = t >> 2, cofs = (t & 3) * 8;
  const bool rok = (tileBase + r) < count;
  const size_t arow = (size_t)(baseE + tileBase + (rok ? r : 0));
  const float*    wbase = w2  + (size_t)e * HID * DIM;
  const unsigned* wpk   = w2p + (size_t)e * (HID / 2) * DIM;

  for (int kb = 0; kb < HID; kb += 32) {
    __syncthreads();
    uint4 av = {0u, 0u, 0u, 0u};
    if (rok) av = *(const uint4*)(act + arow * HID + kb + cofs);
    unsigned* dst = &AsU[r * 18 + (cofs >> 1)];
    dst[0] = av.x; dst[1] = av.y; dst[2] = av.z; dst[3] = av.w;
    __syncthreads();

    FragBF bf;
#pragma unroll
    for (int v = 0; v < 8; ++v) {
      if constexpr (PACKED) {
        const int k2 = (kb >> 1) + v + half * 8;
        bf.u[v] = wpk[(size_t)k2 * DIM + col];
      } else {
        const int k = kb + v * 2 + half * 16;
        bf.u[v] = packbf(wbase[(size_t)k * DIM + col], wbase[(size_t)(k + 1) * DIM + col]);
      }
    }
#pragma unroll
    for (int m = 0; m < 4; ++m) {
      FragBF af;
      const unsigned* src = &AsU[(m * 16 + id) * 18 + half * 8];
#pragma unroll
      for (int v = 0; v < 8; ++v) af.u[v] = src[v];
      acc[m] = __builtin_amdgcn_wmma_f32_16x16x32_bf16(
          false, af.v, false, bf.v, (short)0, acc[m], false, false);
    }
  }

  const float bias = b2[e * DIM + col];
#pragma unroll
  for (int m = 0; m < 4; ++m) {
#pragma unroll
    for (int jj = 0; jj < 8; ++jj) {
      const int lr = m * 16 + jj + half * 8;
      if (tileBase + lr < count) {
        const int entry = rowlist[baseE + tileBase + lr];   // token*2+k
        const float wv = tkval[entry];
        ybuf[(size_t)entry * DIM + col] = wv * (acc[m][jj] + bias);
      }
    }
  }
}

// ---------------- kernel 7: combine + LayerNorm + head ----------------
__global__ __launch_bounds__(256) void moe_ln_head(
    const float* __restrict__ ybuf, const float* __restrict__ ln_g,
    const float* __restrict__ ln_b, const float* __restrict__ head_w,
    const float* __restrict__ head_b, float* __restrict__ out) {
  const int b = blockIdx.x, t = threadIdx.x;
  __shared__ float red[256];
  __shared__ float nrm[DIM];
  const float4 a0 = ((const float4*)(ybuf + (size_t)(b * 2 + 0) * DIM))[t];
  const float4 a1 = ((const float4*)(ybuf + (size_t)(b * 2 + 1) * DIM))[t];
  float4 c; c.x = a0.x + a1.x; c.y = a0.y + a1.y; c.z = a0.z + a1.z; c.w = a0.w + a1.w;

  red[t] = c.x + c.y + c.z + c.w;
  __syncthreads();
  for (int o = 128; o > 0; o >>= 1) { if (t < o) red[t] += red[t + o]; __syncthreads(); }
  const float mu = red[0] * (1.0f / DIM);
  __syncthreads();
  const float dx = c.x - mu, dy = c.y - mu, dz = c.z - mu, dw = c.w - mu;
  red[t] = dx * dx + dy * dy + dz * dz + dw * dw;
  __syncthreads();
  for (int o = 128; o > 0; o >>= 1) { if (t < o) red[t] += red[t + o]; __syncthreads(); }
  const float rs = rsqrtf(red[0] * (1.0f / DIM) + 1e-5f);
  __syncthreads();
  const int d = t * 4;
  nrm[d + 0] = dx * rs * ln_g[d + 0] + ln_b[d + 0];
  nrm[d + 1] = dy * rs * ln_g[d + 1] + ln_b[d + 1];
  nrm[d + 2] = dz * rs * ln_g[d + 2] + ln_b[d + 2];
  nrm[d + 3] = dw * rs * ln_g[d + 3] + ln_b[d + 3];
  __syncthreads();
  if (t < 160) {
    const int cc = t >> 3, j = t & 7;
    float p = 0.f;
    for (int dd = j; dd < DIM; dd += 8) p += nrm[dd] * head_w[dd * 20 + cc];
    red[t] = p;
  }
  __syncthreads();
  if (t < 20) {
    float s = head_b[t];
    for (int j = 0; j < 8; ++j) s += red[t * 8 + j];
    out[(size_t)b * 20 + t] = s;
  }
}

// ---------------- host launcher ----------------
extern "C" void kernel_launch(void* const* d_in, const int* in_sizes, int n_in,
                              void* d_out, int out_size, void* d_ws, size_t ws_size,
                              hipStream_t stream) {
  const int*   x      = (const int*)d_in[0];
  const float* embed  = (const float*)d_in[1];
  const float* gate_w = (const float*)d_in[2];
  const float* gate_b = (const float*)d_in[3];
  const float* w1     = (const float*)d_in[4];
  const float* b1     = (const float*)d_in[5];
  const float* w2     = (const float*)d_in[6];
  const float* b2     = (const float*)d_in[7];
  const float* ln_g   = (const float*)d_in[8];
  const float* ln_b   = (const float*)d_in[9];
  const float* head_w = (const float*)d_in[10];
  const float* head_b = (const float*)d_in[11];

  float* out_logits = (float*)d_out;                     // [B,20] f32
  int*   out_idx    = (int*)d_out + (size_t)B_TOK * 20;  // [B,2] i32 bits

  char* ws = (char*)d_ws;
  float*          h       = (float*)(ws + OFF_H);
  unsigned short* hb      = (unsigned short*)(ws + OFF_HB);
  float*          tkval   = (float*)(ws + OFF_TKV);
  int*            tkidx   = (int*)(ws + OFF_TKI);
  int*            counts  = (int*)(ws + OFF_CNT);
  int*            base    = (int*)(ws + OFF_BASE);
  int*            cursor  = (int*)(ws + OFF_CUR);
  int*            rowlist = (int*)(ws + OFF_RL);
  unsigned short* act     = (unsigned short*)(ws + OFF_ACT);
  float*          ybuf    = (float*)(ws + OFF_Y);
  unsigned*       w1p     = (unsigned*)(ws + OFF_W1P);
  unsigned*       w2p     = (unsigned*)(ws + OFF_W2P);

  const bool packed = (ws_size >= WS_FULL);   // pure function of ws_size -> deterministic

  moe_zero_counts<<<1, 64, 0, stream>>>(counts);
  moe_embed_pool<<<B_TOK, 256, 0, stream>>>(x, embed, h, hb);
  moe_gate_topk<<<B_TOK, 256, 0, stream>>>(h, gate_w, gate_b, tkval, tkidx, counts, out_idx);
  moe_prefix<<<1, 1, 0, stream>>>(counts, base, cursor);
  moe_assign<<<B_TOK / 256, 256, 0, stream>>>(tkidx, base, cursor, rowlist);

  const dim3 g1(HID / 128, B_TOK / 64, NEXP);
  const dim3 g2(DIM / 128, B_TOK / 64, NEXP);
  if (packed) {
    const size_t n1 = (size_t)NEXP * (DIM / 2) * HID;   // u32 elements
    const size_t n2 = (size_t)NEXP * (HID / 2) * DIM;
    moe_pack_w<<<(unsigned)((n1 + 255) / 256), 256, 0, stream>>>(w1, w1p, DIM, HID);
    moe_pack_w<<<(unsigned)((n2 + 255) / 256), 256, 0, stream>>>(w2, w2p, HID, DIM);
    moe_ffn1_wmma<true><<<g1, 256, 0, stream>>>(hb, w1, w1p, b1, counts, base, rowlist, act);
    moe_ffn2_wmma<true><<<g2, 256, 0, stream>>>(act, w2, w2p, b2, tkval, counts, base, rowlist, ybuf);
  } else {
    moe_ffn1_wmma<false><<<g1, 256, 0, stream>>>(hb, w1, w1p, b1, counts, base, rowlist, act);
    moe_ffn2_wmma<false><<<g2, 256, 0, stream>>>(act, w2, w2p, b2, tkval, counts, base, rowlist, ybuf);
  }

  moe_ln_head<<<B_TOK, 256, 0, stream>>>(ybuf, ln_g, ln_b, head_w, head_b, out_logits);
}
